// Flow0Regressor_45835890983478
// MI455X (gfx1250) — matrix operand order repaired and checked by reading, not compiled, not run on hardware
//
#include <hip/hip_runtime.h>

// Problem shape (fixed by the reference)
#define B_   8
#define N_   4096
#define C_   128
#define S_   1024      // npoint//4
#define K_   32        // NSAMPLE
#define CPAD 160       // 131 channels padded to multiple of 32 for WMMA K-loop
#define EPS_ 1e-5f

typedef __attribute__((ext_vector_type(16))) _Float16 v16h;
typedef __attribute__((ext_vector_type(8)))  _Float16 v8h;
typedef __attribute__((ext_vector_type(8)))  float    v8f;

// ---------------------------------------------------------------------------
// Farthest point sampling: one block per batch, deterministic start at idx 0.
// dist kept in global ws (L2-resident: 16 KB/batch), argmax via LDS tree with
// lowest-index tie-break to match jnp.argmax.
// ---------------------------------------------------------------------------
__global__ __launch_bounds__(256)
void fps_kernel(const float* __restrict__ xyz /*[B,3,N]*/,
                int* __restrict__ fps_idx /*[B,S]*/,
                float* __restrict__ dist /*[B,N]*/) {
  const int b = blockIdx.x;
  const int t = threadIdx.x;
  const float* px = xyz + (size_t)b * 3 * N_;
  const float* py = px + N_;
  const float* pz = px + 2 * N_;
  float* db = dist + (size_t)b * N_;

  __shared__ float smax[256];
  __shared__ int   simax[256];
  __shared__ float cent[3];

  for (int n = t; n < N_; n += 256) db[n] = 1e10f;
  __syncthreads();

  int far = 0;
  for (int it = 0; it < S_; ++it) {
    if (t == 0) {
      fps_idx[b * S_ + it] = far;
      cent[0] = px[far]; cent[1] = py[far]; cent[2] = pz[far];
    }
    __syncthreads();
    const float cx = cent[0], cy = cent[1], cz = cent[2];
    float bestd = -1.0f; int besti = 0;
    for (int n = t; n < N_; n += 256) {
      const float dx = px[n] - cx, dy = py[n] - cy, dz = pz[n] - cz;
      const float d = dx * dx + dy * dy + dz * dz;
      const float dd = fminf(db[n], d);
      db[n] = dd;
      if (dd > bestd) { bestd = dd; besti = n; }
    }
    smax[t] = bestd; simax[t] = besti;
    __syncthreads();
    for (int off = 128; off > 0; off >>= 1) {
      if (t < off) {
        if (smax[t + off] > smax[t] ||
            (smax[t + off] == smax[t] && simax[t + off] < simax[t])) {
          smax[t] = smax[t + off]; simax[t] = simax[t + off];
        }
      }
      __syncthreads();
    }
    far = simax[0];
    __syncthreads();
  }
}

// ---------------------------------------------------------------------------
// kNN: one thread per center; insertion-sorted top-32 of squared distances.
// ---------------------------------------------------------------------------
__global__ __launch_bounds__(128)
void knn_kernel(const float* __restrict__ xyz,
                const int* __restrict__ fps_idx,
                int* __restrict__ knn_idx /*[B*S,K]*/,
                float* __restrict__ new_xyz /*[B*S,3]*/) {
  const int gid = blockIdx.x * 128 + threadIdx.x;
  if (gid >= B_ * S_) return;
  const int b = gid / S_;
  const float* px = xyz + (size_t)b * 3 * N_;
  const float* py = px + N_;
  const float* pz = px + 2 * N_;
  const int ci = fps_idx[gid];
  const float cx = px[ci], cy = py[ci], cz = pz[ci];
  new_xyz[gid * 3 + 0] = cx;
  new_xyz[gid * 3 + 1] = cy;
  new_xyz[gid * 3 + 2] = cz;

  float bd[K_]; int bi[K_];
#pragma unroll
  for (int k = 0; k < K_; ++k) { bd[k] = 3.0e38f; bi[k] = 0; }

  for (int n = 0; n < N_; ++n) {
    const float dx = px[n] - cx, dy = py[n] - cy, dz = pz[n] - cz;
    const float d = dx * dx + dy * dy + dz * dz;
    if (d < bd[K_ - 1]) {
      int j = K_ - 1;
      while (j > 0 && bd[j - 1] > d) { bd[j] = bd[j - 1]; bi[j] = bi[j - 1]; --j; }
      bd[j] = d; bi[j] = n;
    }
  }
  int* o = knn_idx + (size_t)gid * K_;
#pragma unroll
  for (int k = 0; k < K_; ++k) o[k] = bi[k];
}

// ---------------------------------------------------------------------------
// Gather grouped features into f16 activation matrix [M, CPAD]:
// ch 0..2 = xyz - center, ch 3..130 = point features, ch 131..159 = 0.
// ---------------------------------------------------------------------------
__global__ __launch_bounds__(256)
void gather_kernel(const float* __restrict__ xyz,
                   const float* __restrict__ feat /*[B,128,N]*/,
                   const int* __restrict__ knn_idx,
                   const float* __restrict__ new_xyz,
                   _Float16* __restrict__ X /*[M,CPAD]*/) {
  const int r = blockIdx.x * 256 + threadIdx.x; // row = ((b*S)+s)*K + k
  if (r >= B_ * S_ * K_) return;
  const int b = r / (S_ * K_);
  const int bs = r / K_;
  const int n = knn_idx[r];
  const float* px = xyz + (size_t)b * 3 * N_;
  _Float16* o = X + (size_t)r * CPAD;
  o[0] = (_Float16)(px[n] - new_xyz[bs * 3 + 0]);
  o[1] = (_Float16)(px[N_ + n] - new_xyz[bs * 3 + 1]);
  o[2] = (_Float16)(px[2 * N_ + n] - new_xyz[bs * 3 + 2]);
  const float* pf = feat + (size_t)b * C_ * N_ + n;
#pragma unroll 4
  for (int c = 0; c < C_; ++c) o[3 + c] = (_Float16)pf[(size_t)c * N_];
#pragma unroll
  for (int c = 3 + C_; c < CPAD; ++c) o[c] = (_Float16)0.f;
}

// ---------------------------------------------------------------------------
// Weight conversion f32[128,inc] -> f16[128,ldw] (zero padded).
// ---------------------------------------------------------------------------
__global__ __launch_bounds__(256)
void convert_w_kernel(const float* __restrict__ W, _Float16* __restrict__ Wh,
                      int inc, int ldw) {
  const int i = blockIdx.x * 256 + threadIdx.x;
  if (i >= 128 * ldw) return;
  const int o = i / ldw, k = i % ldw;
  Wh[i] = (k < inc) ? (_Float16)W[o * inc + k] : (_Float16)0.f;
}

// ---------------------------------------------------------------------------
// WMMA GEMM: Y[M,128] = X[M,ldx(f16)] * W^T + bias.  W is [128, ldw] f16
// row-major; it is staged once per block into LDS (<= 40 KB of the 320 KB
// WGP budget) so B-fragments come from ds_load_b128 instead of global.
// Each wave owns TWO 16-row tiles (32 rows x 128 cols): each B-fragment read
// feeds two v_wmma_f32_16x16x32_f16, doubling WMMA density per load.
//
// A-fragment per ISA 16-bit 16x32 layout:
//   lanes 0-15 : halves 0-7 = K 0-7,  halves 8-15 = K 16-23
//   lanes 16-31: halves 0-7 = K 8-15, halves 8-15 = K 24-31
// B-fragment: lane (n%16) of column n uses the same K layout as A
//   (lanes 0-15 -> K 0-7,16-23; lanes 16-31 -> K 8-15,24-31); W rows are contiguous K.
// D tile: vgpr r -> (M=r, N=lane) lanes 0-15 / (M=r+8, N=lane-16) lanes 16-31.
// ---------------------------------------------------------------------------
__global__ __launch_bounds__(256)
void gemm_bias_wmma(const _Float16* __restrict__ X, int ldx,
                    const _Float16* __restrict__ Wt, int ldw,
                    const float* __restrict__ bias,
                    float* __restrict__ Y, int M, int Kin) {
  __shared__ _Float16 sW[128 * CPAD];  // 40 KB max (ldw <= CPAD)

  // Cooperative stage of the whole weight matrix (contiguous 128*ldw halves).
  {
    const int tot = 128 * ldw;
    for (int i = threadIdx.x * 8; i < tot; i += 256 * 8) {
      *(v8h*)(sW + i) = *(const v8h*)(Wt + i);
    }
  }
  __syncthreads();

  const int lane = threadIdx.x & 31;
  const int wave = threadIdx.x >> 5;
  const int row0 = (blockIdx.x * 16 + wave * 2) * 16;  // 2 row-tiles per wave
  if (row0 >= M) return;            // uniform per wave; EXEC stays all-1
  const int half = lane >> 4;
  const int l16 = lane & 15;

  v8f acc0[8] = {};
  v8f acc1[8] = {};
  const _Float16* arow0 = X + (size_t)(row0 + l16) * ldx + half * 8;
  const _Float16* arow1 = arow0 + (size_t)16 * ldx;

  for (int kc = 0; kc < Kin; kc += 32) {
    const v8h a0lo = *(const v8h*)(arow0 + kc);
    const v8h a0hi = *(const v8h*)(arow0 + kc + 16);
    const v8h a1lo = *(const v8h*)(arow1 + kc);
    const v8h a1hi = *(const v8h*)(arow1 + kc + 16);
    const v16h a0 = __builtin_shufflevector(
        a0lo, a0hi, 0, 1, 2, 3, 4, 5, 6, 7, 8, 9, 10, 11, 12, 13, 14, 15);
    const v16h a1 = __builtin_shufflevector(
        a1lo, a1hi, 0, 1, 2, 3, 4, 5, 6, 7, 8, 9, 10, 11, 12, 13, 14, 15);
#pragma unroll
    for (int nt = 0; nt < 8; ++nt) {
      const _Float16* bp = sW + (nt * 16 + l16) * ldw + kc + half * 8;
      const v16h bfrag = __builtin_shufflevector(*(const v8h*)bp, *(const v8h*)(bp + 16), 0, 1, 2, 3, 4, 5, 6, 7, 8, 9, 10, 11, 12, 13, 14, 15);   // 2x ds_load_b128
      acc0[nt] = __builtin_amdgcn_wmma_f32_16x16x32_f16(
          false, a0, false, bfrag, (short)0, acc0[nt], false, false);
      acc1[nt] = __builtin_amdgcn_wmma_f32_16x16x32_f16(
          false, a1, false, bfrag, (short)0, acc1[nt], false, false);
    }
  }

#pragma unroll
  for (int nt = 0; nt < 8; ++nt) {
    const int n = nt * 16 + l16;
    const float bv = bias[n];
#pragma unroll
    for (int r = 0; r < 8; ++r) {
      const int m = row0 + r + half * 8;
      Y[(size_t)m * 128 + n] = acc0[nt][r] + bv;
      Y[(size_t)(m + 16) * 128 + n] = acc1[nt][r] + bv;
    }
  }
}

// ---------------------------------------------------------------------------
// Instance-norm stats: one block per (b, channel); biased variance.
// stats[(b*128+c)*2] = mean, [..+1] = rsqrt(var + eps)
// ---------------------------------------------------------------------------
__global__ __launch_bounds__(256)
void stats_kernel(const float* __restrict__ Y, float* __restrict__ stats) {
  const int b = blockIdx.x >> 7;
  const int c = blockIdx.x & 127;
  const int SK = S_ * K_;
  const float* p = Y + (size_t)b * SK * 128 + c;
  float s = 0.f, s2 = 0.f;
  for (int i = threadIdx.x; i < SK; i += 256) {
    const float v = p[(size_t)i * 128];
    s += v; s2 += v * v;
  }
  __shared__ float sh[256], sh2[256];
  sh[threadIdx.x] = s; sh2[threadIdx.x] = s2;
  __syncthreads();
  for (int off = 128; off > 0; off >>= 1) {
    if (threadIdx.x < off) {
      sh[threadIdx.x] += sh[threadIdx.x + off];
      sh2[threadIdx.x] += sh2[threadIdx.x + off];
    }
    __syncthreads();
  }
  if (threadIdx.x == 0) {
    const float inv = 1.0f / (float)SK;
    const float mean = sh[0] * inv;
    const float var = sh2[0] * inv - mean * mean;
    stats[(b * 128 + c) * 2 + 0] = mean;
    stats[(b * 128 + c) * 2 + 1] = rsqrtf(var + EPS_);
  }
}

// ---------------------------------------------------------------------------
// Normalize + ReLU, downcast to f16 into the activation buffer (stride CPAD).
// ---------------------------------------------------------------------------
__global__ __launch_bounds__(256)
void norm_relu_kernel(const float* __restrict__ Y,
                      const float* __restrict__ stats,
                      _Float16* __restrict__ X) {
  const size_t total = (size_t)B_ * S_ * K_ * 128;
  const size_t i = (size_t)blockIdx.x * 256 + threadIdx.x;
  if (i >= total) return;
  const int c = (int)(i & 127);
  const size_t row = i >> 7;
  const int b = (int)(row / ((size_t)S_ * K_));
  const float mu = stats[(b * 128 + c) * 2 + 0];
  const float rs = stats[(b * 128 + c) * 2 + 1];
  float v = (Y[i] - mu) * rs;
  v = fmaxf(v, 0.f);
  X[row * CPAD + c] = (_Float16)v;
}

// ---------------------------------------------------------------------------
// Max over K neighbors + final FC(128->3), write flow0 [B,3,S].
// ---------------------------------------------------------------------------
__global__ __launch_bounds__(128)
void maxfc_kernel(const _Float16* __restrict__ X,
                  const float* __restrict__ fcw, const float* __restrict__ fcb,
                  float* __restrict__ out) {
  const int bs = blockIdx.x;     // b*S + s
  const int c = threadIdx.x;     // channel
  const _Float16* p = X + (size_t)bs * K_ * CPAD + c;
  float m = 0.f;                 // post-ReLU values are >= 0
  for (int k = 0; k < K_; ++k) m = fmaxf(m, (float)p[(size_t)k * CPAD]);
  __shared__ float sm[128];
  sm[c] = m;
  __syncthreads();
  if (c < 3) {
    float acc = fcb[c];
#pragma unroll 4
    for (int j = 0; j < 128; ++j) acc += fcw[c * 128 + j] * sm[j];
    const int b = bs / S_, s = bs % S_;
    out[((size_t)b * 3 + c) * S_ + s] = acc;
  }
}

// ---------------------------------------------------------------------------
extern "C" void kernel_launch(void* const* d_in, const int* in_sizes, int n_in,
                              void* d_out, int out_size, void* d_ws, size_t ws_size,
                              hipStream_t stream) {
  const float* pc1  = (const float*)d_in[0];  // [B,3,N]
  const float* corr = (const float*)d_in[1];  // [B,128,N]
  const float* W1   = (const float*)d_in[2];  // [128,131]
  const float* b1   = (const float*)d_in[3];
  const float* W2   = (const float*)d_in[4];  // [128,128]
  const float* b2   = (const float*)d_in[5];
  const float* W3   = (const float*)d_in[6];  // [128,128]
  const float* b3   = (const float*)d_in[7];
  const float* fcw  = (const float*)d_in[8];  // [3,128]
  const float* fcb  = (const float*)d_in[9];
  float* out = (float*)d_out;                 // [B,3,S]

  char* p = (char*)d_ws;
  auto carve = [&](size_t bytes) -> char* {
    char* r = p;
    p += (bytes + 255) & ~(size_t)255;
    return r;
  };
  int*      fps_idx = (int*)carve((size_t)B_ * S_ * 4);
  float*    dist    = (float*)carve((size_t)B_ * N_ * 4);
  int*      knn_idx = (int*)carve((size_t)B_ * S_ * K_ * 4);
  float*    nxyz    = (float*)carve((size_t)B_ * S_ * 3 * 4);
  float*    stats   = (float*)carve((size_t)B_ * 128 * 2 * 4);
  _Float16* Wh1     = (_Float16*)carve((size_t)128 * CPAD * 2);
  _Float16* Wh2     = (_Float16*)carve((size_t)128 * 128 * 2);
  _Float16* Wh3     = (_Float16*)carve((size_t)128 * 128 * 2);
  _Float16* Xb      = (_Float16*)carve((size_t)B_ * S_ * K_ * CPAD * 2); // ~84 MB
  float*    Yb      = (float*)carve((size_t)B_ * S_ * K_ * 128 * 4);     // ~134 MB

  const int M = B_ * S_ * K_;              // 262144 rows
  const size_t total = (size_t)M * 128;

  fps_kernel<<<B_, 256, 0, stream>>>(pc1, fps_idx, dist);
  knn_kernel<<<(B_ * S_ + 127) / 128, 128, 0, stream>>>(pc1, fps_idx, knn_idx, nxyz);
  gather_kernel<<<(M + 255) / 256, 256, 0, stream>>>(pc1, corr, knn_idx, nxyz, Xb);
  convert_w_kernel<<<(128 * CPAD + 255) / 256, 256, 0, stream>>>(W1, Wh1, 131, CPAD);
  convert_w_kernel<<<(128 * 128 + 255) / 256, 256, 0, stream>>>(W2, Wh2, 128, 128);
  convert_w_kernel<<<(128 * 128 + 255) / 256, 256, 0, stream>>>(W3, Wh3, 128, 128);

  // Layer 1: Kin padded 131 -> 160 (zeros in both X and W).
  gemm_bias_wmma<<<M / 256, 256, 0, stream>>>(Xb, CPAD, Wh1, CPAD, b1, Yb, M, CPAD);
  stats_kernel<<<B_ * 128, 256, 0, stream>>>(Yb, stats);
  norm_relu_kernel<<<(int)((total + 255) / 256), 256, 0, stream>>>(Yb, stats, Xb);

  // Layer 2
  gemm_bias_wmma<<<M / 256, 256, 0, stream>>>(Xb, CPAD, Wh2, 128, b2, Yb, M, 128);
  stats_kernel<<<B_ * 128, 256, 0, stream>>>(Yb, stats);
  norm_relu_kernel<<<(int)((total + 255) / 256), 256, 0, stream>>>(Yb, stats, Xb);

  // Layer 3
  gemm_bias_wmma<<<M / 256, 256, 0, stream>>>(Xb, CPAD, Wh3, 128, b3, Yb, M, 128);
  stats_kernel<<<B_ * 128, 256, 0, stream>>>(Yb, stats);
  norm_relu_kernel<<<(int)((total + 255) / 256), 256, 0, stream>>>(Yb, stats, Xb);

  maxfc_kernel<<<B_ * S_, 128, 0, stream>>>(Xb, fcw, fcb, out);

  (void)in_sizes; (void)n_in; (void)out_size; (void)ws_size;
}
